// mp_layer_39943195852850
// MI455X (gfx1250) — compile-verified, matrix operand order checked by reading, not run on hardware
//
#include <hip/hip_runtime.h>
#include <hip/hip_bf16.h>
#include <stdint.h>

// ---------------------------------------------------------------------------
// Problem constants (from the reference)
// ---------------------------------------------------------------------------
#define NN     50000
#define EE     800000
#define CC     128
#define KT_E0  13           // (3C+3=387) padded to 416 -> 13 K-tiles of 32
#define KT_E1  4            // 128/32
#define KT_N0  8            // 256/32
#define KT_N1  4            // 128/32
#define FRAG_HALVES 512     // 32 lanes * 16 bf16 per WMMA fragment

typedef __attribute__((ext_vector_type(16))) __bf16 v16bf;
typedef __attribute__((ext_vector_type(8)))  float  v8f;

union Frag {
    v16bf v;
    uint4 q[2];
    unsigned short h[16];
};

// fp32 -> bf16, round-to-nearest-even
__device__ __forceinline__ unsigned f2bf(float f) {
    unsigned u = __float_as_uint(f);
    u += 0x7FFFu + ((u >> 16) & 1u);
    return u >> 16;
}
__device__ __forceinline__ unsigned pk2(float lo, float hi) {
    return f2bf(lo) | (f2bf(hi) << 16);
}

// Index (in bf16 units) of element (row, K) inside a packed A-fragment region.
// ISA 7.12.2 "16-bit A-Matrix 16x32": lane = row + hi*16 holds
//   halves 0-7  -> K = kt*32 + hi*8 + i
//   halves 8-15 -> K = kt*32 + 16 + hi*8 + (i-8)
__device__ __forceinline__ int afrag_idx(int row, int K) {
    int kt = K >> 5;
    int kl = K & 31;
    int hi = (kl >> 3) & 1;
    int i  = (kl & 7) | ((kl >> 4) << 3);
    return kt * FRAG_HALVES + ((row + (hi << 4)) << 4) + i;
}

// pack 16 floats (two groups of 8 consecutive channels) into one 32B LDS chunk
__device__ __forceinline__ void stage16(unsigned short* dst16, // 16-half slot
                                        const float4 g0a, const float4 g0b,
                                        const float4 g1a, const float4 g1b) {
    uint4 lo, hi;
    lo.x = pk2(g0a.x, g0a.y); lo.y = pk2(g0a.z, g0a.w);
    lo.z = pk2(g0b.x, g0b.y); lo.w = pk2(g0b.z, g0b.w);
    hi.x = pk2(g1a.x, g1a.y); hi.y = pk2(g1a.z, g1a.w);
    hi.z = pk2(g1b.x, g1b.y); hi.w = pk2(g1b.z, g1b.w);
    uint4* d = (uint4*)dst16;
    d[0] = lo; d[1] = hi;
}

// ---------------------------------------------------------------------------
// Weight packing: fp32 [Kreal][128] -> bf16 fragment-packed [kt][n][lane][16]
// B-fragment layout mirrors the A layout with N in place of M.
// ---------------------------------------------------------------------------
__global__ void pack_weights(const float* __restrict__ W, unsigned short* __restrict__ dst,
                             int Kreal, int kt_count) {
    int t = blockIdx.x * blockDim.x + threadIdx.x;
    int total = kt_count * 8 * 32;
    if (t >= total) return;
    int l  = t & 31;
    int f  = t >> 5;          // fragment id = kt*8 + n
    int kt = f >> 3;
    int n  = f & 7;
    int hi  = l >> 4;
    int col = (l & 15) + n * 16;
    unsigned short tmp[16];
#pragma unroll
    for (int i = 0; i < 16; ++i) {
        int K = kt * 32 + ((i < 8) ? (hi * 8 + i) : (16 + hi * 8 + (i - 8)));
        float v = (K < Kreal) ? W[K * CC + col] : 0.0f;
        tmp[i] = (unsigned short)f2bf(v);
    }
    uint4* d = (uint4*)(dst + f * FRAG_HALVES + l * 16);
    d[0] = ((uint4*)tmp)[0];
    d[1] = ((uint4*)tmp)[1];
}

__global__ void zero_f32(float* __restrict__ p, int n) {
    int i = blockIdx.x * blockDim.x + threadIdx.x;
    if (i < n) p[i] = 0.0f;
}

// ---------------------------------------------------------------------------
// Edge kernel: 1 wave = 32 edges (two 16-row A tiles sharing B fragments).
// ea = [x_j, x_i, x_j-x_i, dpos] (K=416); hidden = relu(ea@ew0+eb0);
// out = hidden@ew1+eb1; atomic scatter-add into agg[dst].
// 2 waves/block, 25000 wave-tiles exactly -> 12500 blocks, no tail.
// ---------------------------------------------------------------------------
__global__ __launch_bounds__(64) void edge_kernel(
    const float* __restrict__ x, const int* __restrict__ ei,
    const float* __restrict__ pos,
    const unsigned short* __restrict__ ew0p, const float* __restrict__ eb0,
    const unsigned short* __restrict__ ew1p, const float* __restrict__ eb1,
    float* __restrict__ agg)
{
    __shared__ __align__(16) unsigned short sm[2 * 2 * KT_E0 * FRAG_HALVES]; // 53 KB
    const int wave = threadIdx.x >> 5;
    const int l    = threadIdx.x & 31;
    const int wt   = blockIdx.x * 2 + wave;     // wave-tile: 32 edges
    const int row  = l & 15;
    const int sub  = l >> 4;
    unsigned short* tbase[2];
    tbase[0] = sm + wave * (2 * KT_E0 * FRAG_HALVES);
    tbase[1] = tbase[0] + KT_E0 * FRAG_HALVES;

    // ---- stage both 16-edge tiles into LDS in A-fragment layout ----
#pragma unroll
    for (int t = 0; t < 2; ++t) {
        const int e = wt * 32 + t * 16 + row;
        const int sIdx = ei[e];
        const int dIdx = ei[EE + e];
        const float* xs = x + (size_t)sIdx * CC;
        const float* xo = x + (size_t)dIdx * CC;
        unsigned short* base = tbase[t];

        for (int kt = 0; kt < 12; ++kt) {
            const int part = kt >> 2;            // 0:x_j 1:x_i 2:x_j-x_i
            const int c0 = (kt & 3) * 32 + sub * 8;
            float4 g0a, g0b, g1a, g1b;
            if (part == 0) {
                g0a = *(const float4*)(xs + c0);      g0b = *(const float4*)(xs + c0 + 4);
                g1a = *(const float4*)(xs + c0 + 16); g1b = *(const float4*)(xs + c0 + 20);
            } else if (part == 1) {
                g0a = *(const float4*)(xo + c0);      g0b = *(const float4*)(xo + c0 + 4);
                g1a = *(const float4*)(xo + c0 + 16); g1b = *(const float4*)(xo + c0 + 20);
            } else {
                float4 sa = *(const float4*)(xs + c0),      sb = *(const float4*)(xs + c0 + 4);
                float4 sc = *(const float4*)(xs + c0 + 16), sd = *(const float4*)(xs + c0 + 20);
                float4 oa = *(const float4*)(xo + c0),      ob = *(const float4*)(xo + c0 + 4);
                float4 oc = *(const float4*)(xo + c0 + 16), od = *(const float4*)(xo + c0 + 20);
                g0a = make_float4(sa.x-oa.x, sa.y-oa.y, sa.z-oa.z, sa.w-oa.w);
                g0b = make_float4(sb.x-ob.x, sb.y-ob.y, sb.z-ob.z, sb.w-ob.w);
                g1a = make_float4(sc.x-oc.x, sc.y-oc.y, sc.z-oc.z, sc.w-oc.w);
                g1b = make_float4(sd.x-od.x, sd.y-od.y, sd.z-od.z, sd.w-od.w);
            }
            stage16(base + kt * FRAG_HALVES + l * 16, g0a, g0b, g1a, g1b);
        }
        // kt = 12: dpos (K 384..386) + zero pad
        {
            unsigned short tmp[16];
#pragma unroll
            for (int i = 0; i < 16; ++i) tmp[i] = 0;
            if (sub == 0) {
                tmp[0] = (unsigned short)f2bf(pos[(size_t)sIdx*3+0] - pos[(size_t)dIdx*3+0]);
                tmp[1] = (unsigned short)f2bf(pos[(size_t)sIdx*3+1] - pos[(size_t)dIdx*3+1]);
                tmp[2] = (unsigned short)f2bf(pos[(size_t)sIdx*3+2] - pos[(size_t)dIdx*3+2]);
            }
            uint4* d = (uint4*)(base + 12 * FRAG_HALVES + l * 16);
            d[0] = ((uint4*)tmp)[0];
            d[1] = ((uint4*)tmp)[1];
        }
    }

    // ---- layer 0: 13 K-tiles, B shared by both A tiles (16 WMMA / K-tile) ----
    v8f acc0[8], acc1[8];
#pragma unroll
    for (int n = 0; n < 8; ++n)
#pragma unroll
        for (int j = 0; j < 8; ++j) { acc0[n][j] = 0.0f; acc1[n][j] = 0.0f; }

    for (int kt = 0; kt < KT_E0; ++kt) {
        Frag A0, A1, B[8];
        const uint4* a0p = (const uint4*)(tbase[0] + kt * FRAG_HALVES + l * 16);
        const uint4* a1p = (const uint4*)(tbase[1] + kt * FRAG_HALVES + l * 16);
        A0.q[0] = a0p[0]; A0.q[1] = a0p[1];
        A1.q[0] = a1p[0]; A1.q[1] = a1p[1];
#pragma unroll
        for (int n = 0; n < 8; ++n) {
            const uint4* bp = (const uint4*)(ew0p + (kt * 8 + n) * FRAG_HALVES + l * 16);
            B[n].q[0] = bp[0]; B[n].q[1] = bp[1];
        }
#pragma unroll
        for (int n = 0; n < 8; ++n) {
            acc0[n] = __builtin_amdgcn_wmma_f32_16x16x32_bf16(
                false, A0.v, false, B[n].v, (short)0, acc0[n], false, false);
            acc1[n] = __builtin_amdgcn_wmma_f32_16x16x32_bf16(
                false, A1.v, false, B[n].v, (short)0, acc1[n], false, false);
        }
    }

    // ---- bias + relu, restage hidden (2 x 16x128) into LDS A-frag layout ----
#pragma unroll
    for (int n = 0; n < 8; ++n) {
        int ncol = n * 16 + row;
        float bias = eb0[ncol];
#pragma unroll
        for (int j = 0; j < 8; ++j) {
            int m = j + sub * 8;
            float v0 = acc0[n][j] + bias; v0 = v0 > 0.0f ? v0 : 0.0f;
            float v1 = acc1[n][j] + bias; v1 = v1 > 0.0f ? v1 : 0.0f;
            tbase[0][afrag_idx(m, ncol)] = (unsigned short)f2bf(v0);
            tbase[1][afrag_idx(m, ncol)] = (unsigned short)f2bf(v1);
        }
    }

    // ---- layer 1: 4 K-tiles ----
#pragma unroll
    for (int n = 0; n < 8; ++n)
#pragma unroll
        for (int j = 0; j < 8; ++j) { acc0[n][j] = 0.0f; acc1[n][j] = 0.0f; }

    for (int kt = 0; kt < KT_E1; ++kt) {
        Frag A0, A1, B[8];
        const uint4* a0p = (const uint4*)(tbase[0] + kt * FRAG_HALVES + l * 16);
        const uint4* a1p = (const uint4*)(tbase[1] + kt * FRAG_HALVES + l * 16);
        A0.q[0] = a0p[0]; A0.q[1] = a0p[1];
        A1.q[0] = a1p[0]; A1.q[1] = a1p[1];
#pragma unroll
        for (int n = 0; n < 8; ++n) {
            const uint4* bp = (const uint4*)(ew1p + (kt * 8 + n) * FRAG_HALVES + l * 16);
            B[n].q[0] = bp[0]; B[n].q[1] = bp[1];
        }
#pragma unroll
        for (int n = 0; n < 8; ++n) {
            acc0[n] = __builtin_amdgcn_wmma_f32_16x16x32_bf16(
                false, A0.v, false, B[n].v, (short)0, acc0[n], false, false);
            acc1[n] = __builtin_amdgcn_wmma_f32_16x16x32_bf16(
                false, A1.v, false, B[n].v, (short)0, acc1[n], false, false);
        }
    }

    // ---- bias + atomic scatter-add into agg[dst] ----
#pragma unroll
    for (int t = 0; t < 2; ++t) {
#pragma unroll
        for (int n = 0; n < 8; ++n) {
            int ncol = n * 16 + row;
            float bias = eb1[ncol];
#pragma unroll
            for (int j = 0; j < 8; ++j) {
                int m = j + sub * 8;
                int d = ei[EE + wt * 32 + t * 16 + m];
                float v = (t == 0 ? acc0[n][j] : acc1[n][j]) + bias;
                atomicAdd(&agg[(size_t)d * CC + ncol], v);
            }
        }
    }
}

// ---------------------------------------------------------------------------
// Node kernel: 1 wave = 16 nodes. h = [x, agg] (K=256),
// out = relu(h@nw0+nb0)@nw1 + nb1
// ---------------------------------------------------------------------------
__global__ __launch_bounds__(128) void node_kernel(
    const float* __restrict__ x, const float* __restrict__ agg,
    const unsigned short* __restrict__ nw0p, const float* __restrict__ nb0,
    const unsigned short* __restrict__ nw1p, const float* __restrict__ nb1,
    float* __restrict__ out, int ntiles)
{
    __shared__ __align__(16) unsigned short sm[4 * KT_N0 * FRAG_HALVES]; // 32 KB
    const int wave = threadIdx.x >> 5;
    const int l    = threadIdx.x & 31;
    const int tile = blockIdx.x * 4 + wave;
    if (tile >= ntiles) return;          // no block barriers used -> safe
    unsigned short* base = sm + wave * (KT_N0 * FRAG_HALVES);

    const int row  = l & 15;
    const int sub  = l >> 4;
    const int node = tile * 16 + row;
    const float* xr = x   + (size_t)node * CC;
    const float* ar = agg + (size_t)node * CC;

    // ---- stage h = [x, agg] (vectorized) ----
    for (int kt = 0; kt < 8; ++kt) {
        const float* src = (kt < 4) ? xr : ar;
        const int c0 = (kt & 3) * 32 + sub * 8;
        float4 g0a = *(const float4*)(src + c0);
        float4 g0b = *(const float4*)(src + c0 + 4);
        float4 g1a = *(const float4*)(src + c0 + 16);
        float4 g1b = *(const float4*)(src + c0 + 20);
        stage16(base + kt * FRAG_HALVES + l * 16, g0a, g0b, g1a, g1b);
    }

    v8f acc[8];
#pragma unroll
    for (int n = 0; n < 8; ++n)
#pragma unroll
        for (int j = 0; j < 8; ++j) acc[n][j] = 0.0f;

    for (int kt = 0; kt < KT_N0; ++kt) {
        Frag A, B[8];
        const uint4* ap = (const uint4*)(base + kt * FRAG_HALVES + l * 16);
        A.q[0] = ap[0]; A.q[1] = ap[1];
#pragma unroll
        for (int n = 0; n < 8; ++n) {
            const uint4* bp = (const uint4*)(nw0p + (kt * 8 + n) * FRAG_HALVES + l * 16);
            B[n].q[0] = bp[0]; B[n].q[1] = bp[1];
        }
#pragma unroll
        for (int n = 0; n < 8; ++n)
            acc[n] = __builtin_amdgcn_wmma_f32_16x16x32_bf16(
                false, A.v, false, B[n].v, (short)0, acc[n], false, false);
    }

#pragma unroll
    for (int n = 0; n < 8; ++n) {
        int ncol = n * 16 + row;
        float bias = nb0[ncol];
#pragma unroll
        for (int j = 0; j < 8; ++j) {
            float v = acc[n][j] + bias;
            v = v > 0.0f ? v : 0.0f;
            int m = j + sub * 8;
            base[afrag_idx(m, ncol)] = (unsigned short)f2bf(v);
        }
    }

#pragma unroll
    for (int n = 0; n < 8; ++n)
#pragma unroll
        for (int j = 0; j < 8; ++j) acc[n][j] = 0.0f;

    for (int kt = 0; kt < KT_N1; ++kt) {
        Frag A, B[8];
        const uint4* ap = (const uint4*)(base + kt * FRAG_HALVES + l * 16);
        A.q[0] = ap[0]; A.q[1] = ap[1];
#pragma unroll
        for (int n = 0; n < 8; ++n) {
            const uint4* bp = (const uint4*)(nw1p + (kt * 8 + n) * FRAG_HALVES + l * 16);
            B[n].q[0] = bp[0]; B[n].q[1] = bp[1];
        }
#pragma unroll
        for (int n = 0; n < 8; ++n)
            acc[n] = __builtin_amdgcn_wmma_f32_16x16x32_bf16(
                false, A.v, false, B[n].v, (short)0, acc[n], false, false);
    }

#pragma unroll
    for (int n = 0; n < 8; ++n) {
        int ncol = n * 16 + row;
        float bias = nb1[ncol];
#pragma unroll
        for (int j = 0; j < 8; ++j) {
            int m = j + sub * 8;
            out[(size_t)(tile * 16 + m) * CC + ncol] = acc[n][j] + bias;
        }
    }
}

// ---------------------------------------------------------------------------
// Launch
// ---------------------------------------------------------------------------
extern "C" void kernel_launch(void* const* d_in, const int* in_sizes, int n_in,
                              void* d_out, int out_size, void* d_ws, size_t ws_size,
                              hipStream_t stream) {
    (void)in_sizes; (void)n_in; (void)out_size; (void)ws_size;
    const float* x   = (const float*)d_in[0];
    const int*   ei  = (const int*)d_in[1];
    const float* pos = (const float*)d_in[2];
    const float* ew0 = (const float*)d_in[3];
    const float* eb0 = (const float*)d_in[4];
    const float* ew1 = (const float*)d_in[5];
    const float* eb1 = (const float*)d_in[6];
    const float* nw0 = (const float*)d_in[7];
    const float* nb0 = (const float*)d_in[8];
    const float* nw1 = (const float*)d_in[9];
    const float* nb1 = (const float*)d_in[10];
    float* out = (float*)d_out;

    // workspace layout
    float* agg = (float*)d_ws;                                   // N*C f32
    unsigned short* wbuf = (unsigned short*)((char*)d_ws + (size_t)NN * CC * 4);
    unsigned short* ew0p = wbuf;                                 // 13*8*512 halves
    unsigned short* ew1p = ew0p + KT_E0 * 8 * FRAG_HALVES;       //  4*8*512
    unsigned short* nw0p = ew1p + KT_E1 * 8 * FRAG_HALVES;       //  8*8*512
    unsigned short* nw1p = nw0p + KT_N0 * 8 * FRAG_HALVES;       //  4*8*512

    // 1) zero aggregation buffer (every launch; graph-replay safe)
    {
        int n = NN * CC;
        zero_f32<<<(n + 255) / 256, 256, 0, stream>>>(agg, n);
    }
    // 2) pack weights to bf16 fragments
    {
        int t0 = KT_E0 * 8 * 32;
        pack_weights<<<(t0 + 127) / 128, 128, 0, stream>>>(ew0, ew0p, 3 * CC + 3, KT_E0);
        int t1 = KT_E1 * 8 * 32;
        pack_weights<<<(t1 + 127) / 128, 128, 0, stream>>>(ew1, ew1p, CC, KT_E1);
        int t2 = KT_N0 * 8 * 32;
        pack_weights<<<(t2 + 127) / 128, 128, 0, stream>>>(nw0, nw0p, 2 * CC, KT_N0);
        int t3 = KT_N1 * 8 * 32;
        pack_weights<<<(t3 + 127) / 128, 128, 0, stream>>>(nw1, nw1p, CC, KT_N1);
    }
    // 3) edge MLP + scatter-add: 25000 wave-tiles of 32 edges, 2 waves/block
    edge_kernel<<<EE / 32 / 2, 64, 0, stream>>>(x, ei, pos, ew0p, eb0, ew1p, eb1, agg);
    // 4) node MLP: 3125 tiles of 16 nodes, 4 waves/block
    {
        int ntiles = NN / 16;  // 3125
        node_kernel<<<(ntiles + 3) / 4, 128, 0, stream>>>(x, agg, nw0p, nb0, nw1p, nb1, out, ntiles);
    }
}
